// MiniMaxSoftmaxAttention_27101243638160
// MI455X (gfx1250) — matrix operand order repaired and checked by reading, not compiled, not run on hardware
//
#include <hip/hip_runtime.h>

// ---------------------------------------------------------------------------
// CDNA5 (gfx1250) attention layer: QKV GEMM + partial RoPE + flash attention
// + output GEMM, all matmuls via v_wmma_f32_16x16x32_bf16 (wave32).
// Double-buffered LDS staging with async global->LDS copies (ASYNCcnt),
// v_perm-based transposes, hardware bf16 converts, next-tile prefetch.
// GEMM wave tile 64x64: 16 WMMAs per K-step per wave for 8 fragment loads.
// ---------------------------------------------------------------------------

typedef __bf16 v16bf __attribute__((ext_vector_type(16)));
typedef __bf16 v2bf  __attribute__((ext_vector_type(2)));
typedef float  v8f   __attribute__((ext_vector_type(8)));
typedef float  v2f   __attribute__((ext_vector_type(2)));

union FragB16 {            // 16 bf16 = 32 bytes = 8 VGPRs
    v16bf v;
    uint4 u[2];
};

static __device__ inline float bf2f(unsigned short u) {
    unsigned int x = ((unsigned int)u) << 16;
    float f;
    __builtin_memcpy(&f, &x, 4);
    return f;
}

// native RNE converts -> v_cvt_pk_bf16_f32 where available
static __device__ inline unsigned short f2bf(float f) {
    __bf16 b = (__bf16)f;
    unsigned short r;
    __builtin_memcpy(&r, &b, 2);
    return r;
}

static __device__ inline unsigned pack_bf16(float lo, float hi) {
    v2f f = {lo, hi};
    v2bf b = __builtin_convertvector(f, v2bf);
    unsigned r;
    __builtin_memcpy(&r, &b, 4);
    return r;
}

static __device__ inline v8f wmma_bf16(const FragB16& a, const FragB16& b, v8f c) {
    // 8-arg form: (neg_a, A, neg_b, B, c_mod, C, reuse_a, reuse_b)
    return __builtin_amdgcn_wmma_f32_16x16x32_bf16(false, a.v, false, b.v,
                                                   (short)0, c, false, false);
}

// LDS byte offset for async ops (hardware adds the wave's LDS base)
typedef unsigned short __attribute__((address_space(3))) *lds_us_ptr;
static __device__ inline unsigned lds_off(void* p) {
    return (unsigned)(unsigned long long)(lds_us_ptr)p;
}

// ---------------------------------------------------------------------------
// Tiled GEMM:  C[M,N] = A[M,K] * B[K,N]
// Block tile 256x128x32, 256 threads = 8 waves in a 4(M) x 2(N) wave grid,
// each wave computes 64x64 via 4x4 WMMA accumulators (16 WMMA / K-step for
// 8 fragment loads). Double-buffered LDS: tile k+1 staged (async b128 copies
// for bf16 A, packed ds_store_b32 k-pairs for B) while tile k feeds the
// WMMAs; one barrier and one s_wait_asynccnt per K-step. Row stride 40
// halves = 80 B keeps 16B alignment and conflict-free b128 fragment reads.
// ---------------------------------------------------------------------------
template <bool AF32, bool BF32, bool OUTF32>
__global__ void __launch_bounds__(256)
gemm_wmma(const void* __restrict__ Ap, const void* __restrict__ Bp,
          unsigned short* __restrict__ Cb, float* __restrict__ Cf,
          int M, int N, int K, int lda, int ldb, int ldc)
{
    constexpr int BM = 256, BN = 128, BK = 32, AST = 40;
    __shared__ __align__(16) unsigned short As[2][BM * AST];
    __shared__ __align__(16) unsigned short Bs[2][BN * AST];   // [n][k]

    const int tid  = threadIdx.x;
    const int lane = tid & 31;
    const int wave = tid >> 5;
    const int wr   = wave >> 1;     // 0..3 -> M sub-tile (64 rows)
    const int wc   = wave & 1;      // 0..1 -> N sub-tile (64 cols)
    const int m0   = blockIdx.y * BM;
    const int n0   = blockIdx.x * BN;

    v8f acc[4][4] = {};

    const int arow = tid;                                   // A: one row each
    const int bkp  = (tid >> 4) << 1, bnh = (tid & 15) * 8; // B: 2 k-rows x 8 cols

    auto stage = [&](int buf, int k0) {
        // ---- A tile: 256 rows x 32 k, one row (64 B) per thread ----
        if (AF32) {
            const float* A = (const float*)Ap + (size_t)(m0 + arow) * lda + k0;
            unsigned tmp[16];
#pragma unroll
            for (int i = 0; i < 8; ++i) {
                float4 f = *(const float4*)(A + 4 * i);
                tmp[2 * i + 0] = pack_bf16(f.x, f.y);
                tmp[2 * i + 1] = pack_bf16(f.z, f.w);
            }
#pragma unroll
            for (int i = 0; i < 4; ++i)
                *(uint4*)&As[buf][arow * AST + 8 * i] = *(uint4*)&tmp[4 * i];
            if (k0 + BK < K) __builtin_prefetch(A + BK, 0, 3);
        } else {
            // CDNA5 async copy: global -> LDS without touching VGPRs (ASYNCcnt)
            const unsigned short* A =
                (const unsigned short*)Ap + (size_t)(m0 + arow) * lda + k0;
            unsigned l0 = lds_off(&As[buf][arow * AST]);
#pragma unroll
            for (int i = 0; i < 4; ++i)
                asm volatile("global_load_async_to_lds_b128 %0, %1, off"
                             :: "v"(l0 + 16 * i),
                                "v"((unsigned long long)(A + 8 * i)) : "memory");
            if (k0 + BK < K) __builtin_prefetch(A + BK, 0, 3);
        }
        // ---- B tile, transposed, packed k-pairs ----
        if (BF32) {
            const float* B0 = (const float*)Bp + (size_t)(k0 + bkp) * ldb + n0 + bnh;
            const float* B1 = B0 + ldb;
            float4 a0 = *(const float4*)(B0), a1 = *(const float4*)(B0 + 4);
            float4 c0 = *(const float4*)(B1), c1 = *(const float4*)(B1 + 4);
            *(unsigned*)&Bs[buf][(bnh + 0) * AST + bkp] = pack_bf16(a0.x, c0.x);
            *(unsigned*)&Bs[buf][(bnh + 1) * AST + bkp] = pack_bf16(a0.y, c0.y);
            *(unsigned*)&Bs[buf][(bnh + 2) * AST + bkp] = pack_bf16(a0.z, c0.z);
            *(unsigned*)&Bs[buf][(bnh + 3) * AST + bkp] = pack_bf16(a0.w, c0.w);
            *(unsigned*)&Bs[buf][(bnh + 4) * AST + bkp] = pack_bf16(a1.x, c1.x);
            *(unsigned*)&Bs[buf][(bnh + 5) * AST + bkp] = pack_bf16(a1.y, c1.y);
            *(unsigned*)&Bs[buf][(bnh + 6) * AST + bkp] = pack_bf16(a1.z, c1.z);
            *(unsigned*)&Bs[buf][(bnh + 7) * AST + bkp] = pack_bf16(a1.w, c1.w);
            if (k0 + BK < K) __builtin_prefetch(B0 + (size_t)BK * ldb, 0, 3);
        } else {
            const unsigned short* B0 =
                (const unsigned short*)Bp + (size_t)(k0 + bkp) * ldb + n0 + bnh;
            const unsigned short* B1 = B0 + ldb;
#pragma unroll
            for (int j = 0; j < 8; ++j)
                *(unsigned*)&Bs[buf][(bnh + j) * AST + bkp] =
                    __builtin_amdgcn_perm((unsigned)B1[j], (unsigned)B0[j], 0x05040100u);
        }
    };

    const int nT = K / BK;
    stage(0, 0);
    if (!AF32) asm volatile("s_wait_asynccnt 0x0" ::: "memory");
    __syncthreads();

    for (int kt = 0; kt < nT; ++kt) {
        const int cur = kt & 1;
        if (kt + 1 < nT) stage(cur ^ 1, (kt + 1) * BK);

        // ---- WMMA fragments + 16 matrix ops on buffer `cur` ----
        FragB16 af[4];
#pragma unroll
        for (int fm = 0; fm < 4; ++fm) {
            // A-layout (16x32 bf16): lanes 0-15 hold K 0..7 & 16..23,
            // lanes 16-31 hold K 8..15 & 24..31.
            const int row = wr * 64 + fm * 16 + (lane & 15);
            const int klo = (lane >> 4) * 8;
            af[fm].u[0] = *(const uint4*)&As[cur][row * AST + klo];
            af[fm].u[1] = *(const uint4*)&As[cur][row * AST + klo + 16];
        }
#pragma unroll
        for (int fn = 0; fn < 4; ++fn) {
            // B-layout (32x16 bf16): lane n = column, K 0..7 & 16..23 (lanes 0-15)
            // or K 8..15 & 24..31 (lanes 16-31), matching the A layout.
            const int col = wc * 64 + fn * 16 + (lane & 15);
            const int klo = (lane >> 4) * 8;
            FragB16 bf;
            bf.u[0] = *(const uint4*)&Bs[cur][col * AST + klo];
            bf.u[1] = *(const uint4*)&Bs[cur][col * AST + klo + 16];
#pragma unroll
            for (int fm = 0; fm < 4; ++fm)
                acc[fm][fn] = wmma_bf16(af[fm], bf, acc[fm][fn]);
        }

        if (!AF32) asm volatile("s_wait_asynccnt 0x0" ::: "memory");
        __syncthreads();
    }

    // ---- epilogue: C-layout (lane = column, VGPR r = row r / r+8) ----
#pragma unroll
    for (int fm = 0; fm < 4; ++fm)
#pragma unroll
        for (int fn = 0; fn < 4; ++fn)
#pragma unroll
            for (int r = 0; r < 8; ++r) {
                const int row = m0 + wr * 64 + fm * 16 + ((lane >> 4) << 3) + r;
                const int col = n0 + wc * 64 + fn * 16 + (lane & 15);
                if (OUTF32) Cf[(size_t)row * ldc + col] = acc[fm][fn][r];
                else        Cb[(size_t)row * ldc + col] = f2bf(acc[fm][fn][r]);
            }
}

// ---------------------------------------------------------------------------
// Partial RoPE (rotary_dim=64) applied in place on the bf16 qkv buffer.
// Folds the 1/sqrt(128) attention scale into Q. Thread owns a (d, d+32) pair
// so in-place update is race-free. K heads rotated without scale, V untouched.
// ---------------------------------------------------------------------------
__global__ void __launch_bounds__(256)
rope_scale(unsigned short* __restrict__ qkv, const int* __restrict__ positions)
{
    const int LD = 6144;
    const int row = blockIdx.x;
    const float pos = (float)positions[row];
    unsigned short* base = qkv + (size_t)row * LD;
    const float qscale = 0.08838834764831845f;          // 128^-0.5

    for (int u = threadIdx.x; u < 3328; u += 256) {
        if (u < 1280) {                                  // 40 heads x 32 pairs
            const int head = u >> 5, i = u & 31;
            const int col = (head < 32) ? head * 128 + i
                                        : 4096 + (head - 32) * 128 + i;
            const float x1 = bf2f(base[col]);
            const float x2 = bf2f(base[col + 32]);
            const float invf = __powf(1.0e7f, -(float)i * (1.0f / 32.0f));
            const float ang = pos * invf;
            float s, c;
            __sincosf(ang, &s, &c);
            float o1 = x1 * c - x2 * s;
            float o2 = x2 * c + x1 * s;
            if (head < 32) { o1 *= qscale; o2 *= qscale; }
            base[col]      = f2bf(o1);
            base[col + 32] = f2bf(o2);
        } else {                                         // Q pass-through dims
            const int e = u - 1280;
            const int head = e >> 6, d = 64 + (e & 63);
            const int col = head * 128 + d;
            base[col] = f2bf(bf2f(base[col]) * qscale);
        }
    }
}

// ---------------------------------------------------------------------------
// Flash attention, causal, GQA (32 q-heads over 8 kv-heads).
// grid = (T/128 q-tiles, 32 heads), 256 threads = 8 waves, wave owns 16 rows.
// Q fragments in registers; K B-fragments straight from global (contiguous
// per lane); V staged transposed into per-wave LDS with v_perm halfword
// merges + packed b64 stores (4 keys per lane); P transposed C->A layout
// through per-wave LDS scratch. Online softmax via 16-lane __shfl_xor.
// ---------------------------------------------------------------------------
__global__ void __launch_bounds__(256)
fattn(const unsigned short* __restrict__ qkv, unsigned short* __restrict__ attn)
{
    extern __shared__ __align__(16) unsigned short smem[];
    const int LD = 6144;
    const int tid = threadIdx.x, lane = tid & 31, wave = tid >> 5;
    const int h = blockIdx.y, kvh = h >> 2;
    const int qt = blockIdx.x * 128 + wave * 16;

    unsigned short* VT = smem + wave * 4608;   // [128 hd][32 keys] transposed
    unsigned short* PB = VT + 4096;            // [16 rows][32 keys] P scratch

    // Q A-fragments (16 rows x 128 dims = 4 frags of K=32), pre-scaled by rope
    FragB16 qa[4];
    {
        const unsigned short* qb =
            qkv + (size_t)(qt + (lane & 15)) * LD + h * 128 + ((lane >> 4) * 8);
#pragma unroll
        for (int s = 0; s < 4; ++s) {
            qa[s].u[0] = *(const uint4*)(qb + 32 * s);
            qa[s].u[1] = *(const uint4*)(qb + 32 * s + 16);
        }
    }

    v8f o[8] = {};
    float mi[8], li[8];
#pragma unroll
    for (int r = 0; r < 8; ++r) { mi[r] = -3.0e30f; li[r] = 0.0f; }

    const int nb = (qt + 16 + 31) >> 5;        // causal: keys <= qt+15
    for (int b = 0; b < nb; ++b) {
        const int kb = b * 32;

        // ---- S = Q K^T for 32 keys (two 16-col fragments): issue K loads
        //      first, they are on the critical path.
        v8f s0 = {}, s1 = {};
        const unsigned short* kbase =
            qkv + 4096 + (size_t)kvh * 128 + ((lane >> 4) * 8);
#pragma unroll
        for (int s = 0; s < 4; ++s) {
            FragB16 k0f, k1f;
            const unsigned short* p0 =
                kbase + (size_t)(kb + (lane & 15)) * LD + 32 * s;
            const unsigned short* p1 =
                kbase + (size_t)(kb + 16 + (lane & 15)) * LD + 32 * s;
            k0f.u[0] = *(const uint4*)p0; k0f.u[1] = *(const uint4*)(p0 + 16);
            k1f.u[0] = *(const uint4*)p1; k1f.u[1] = *(const uint4*)(p1 + 16);
            s0 = wmma_bf16(qa[s], k0f, s0);
            s1 = wmma_bf16(qa[s], k1f, s1);
        }

        // ---- stage V^T (consumed after softmax): lane owns 4 keys x 32 dims,
        //      v_perm halfword merges -> packed 8B stores.
        {
            const int g = lane >> 2;           // keys 4g..4g+3
            const int c = lane & 3;            // dims 32c..32c+31
            const unsigned short* v0 =
                qkv + (size_t)(kb + 4 * g) * LD + 5120 + kvh * 128 + 32 * c;
            const unsigned short* v1 = v0 + LD;
            const unsigned short* v2 = v1 + LD;
            const unsigned short* v3 = v2 + LD;
#pragma unroll
            for (int i = 0; i < 4; ++i) {      // 8 dims per iteration
                uint4 w0 = *(const uint4*)(v0 + i * 8);
                uint4 w1 = *(const uint4*)(v1 + i * 8);
                uint4 w2 = *(const uint4*)(v2 + i * 8);
                uint4 w3 = *(const uint4*)(v3 + i * 8);
                const unsigned* a0 = (const unsigned*)&w0;
                const unsigned* a1 = (const unsigned*)&w1;
                const unsigned* a2 = (const unsigned*)&w2;
                const unsigned* a3 = (const unsigned*)&w3;
#pragma unroll
                for (int cc = 0; cc < 4; ++cc) {   // 2 dims per component
                    const int d = 32 * c + i * 8 + 2 * cc;
                    uint2 ev, od;
                    ev.x = __builtin_amdgcn_perm(a1[cc], a0[cc], 0x05040100u);
                    ev.y = __builtin_amdgcn_perm(a3[cc], a2[cc], 0x05040100u);
                    od.x = __builtin_amdgcn_perm(a1[cc], a0[cc], 0x07060302u);
                    od.y = __builtin_amdgcn_perm(a3[cc], a2[cc], 0x07060302u);
                    *(uint2*)&VT[(d + 0) * 32 + 4 * g] = ev;
                    *(uint2*)&VT[(d + 1) * 32 + 4 * g] = od;
                }
            }
        }

        // ---- causal mask + online softmax (rows live across 16-lane groups)
        const int rowb = qt + ((lane >> 4) << 3);
        const int kc0 = kb + (lane & 15), kc1 = kc0 + 16;
#pragma unroll
        for (int r = 0; r < 8; ++r) {
            if (kc0 > rowb + r) s0[r] = -1.0e30f;
            if (kc1 > rowb + r) s1[r] = -1.0e30f;
            float v = fmaxf(s0[r], s1[r]);
            v = fmaxf(v, __shfl_xor(v, 1));
            v = fmaxf(v, __shfl_xor(v, 2));
            v = fmaxf(v, __shfl_xor(v, 4));
            v = fmaxf(v, __shfl_xor(v, 8));
            const float mn = fmaxf(mi[r], v);
            const float al = __expf(mi[r] - mn);
            mi[r] = mn;
            const float p0 = __expf(s0[r] - mn);
            const float p1 = __expf(s1[r] - mn);
            float sum = p0 + p1;
            sum += __shfl_xor(sum, 1);
            sum += __shfl_xor(sum, 2);
            sum += __shfl_xor(sum, 4);
            sum += __shfl_xor(sum, 8);
            li[r] = li[r] * al + sum;
#pragma unroll
            for (int j = 0; j < 8; ++j) o[j][r] *= al;
            const int prow = ((lane >> 4) << 3) + r;
            PB[prow * 32 + (lane & 15)]      = f2bf(p0);
            PB[prow * 32 + 16 + (lane & 15)] = f2bf(p1);
        }

        // intra-wave LDS RAW: DS is in-order per wave; wait + fence compiler
        asm volatile("s_wait_dscnt 0" ::: "memory");

        // ---- O += P V : P as A-fragment, V^T rows as B-fragments ----
        FragB16 pa;
        {
            const int prow = lane & 15, klo = (lane >> 4) * 8;
            pa.u[0] = *(const uint4*)&PB[prow * 32 + klo];
            pa.u[1] = *(const uint4*)&PB[prow * 32 + klo + 16];
        }
#pragma unroll
        for (int j = 0; j < 8; ++j) {
            FragB16 vb;
            const int hd = 16 * j + (lane & 15), klo = (lane >> 4) * 8;
            vb.u[0] = *(const uint4*)&VT[hd * 32 + klo];
            vb.u[1] = *(const uint4*)&VT[hd * 32 + klo + 16];
            o[j] = wmma_bf16(pa, vb, o[j]);
        }
    }

    // ---- epilogue: normalize and store bf16 [T, 32*128] ----
#pragma unroll
    for (int j = 0; j < 8; ++j) {
#pragma unroll
        for (int r = 0; r < 8; ++r) {
            const int row = qt + ((lane >> 4) << 3) + r;
            const int col = h * 128 + 16 * j + (lane & 15);
            attn[(size_t)row * 4096 + col] = f2bf(o[j][r] / li[r]);
        }
    }
}

// ---------------------------------------------------------------------------
extern "C" void kernel_launch(void* const* d_in, const int* in_sizes, int n_in,
                              void* d_out, int out_size, void* d_ws, size_t ws_size,
                              hipStream_t stream)
{
    (void)in_sizes; (void)n_in; (void)out_size; (void)ws_size;

    const int*   positions = (const int*)d_in[0];
    const float* hidden    = (const float*)d_in[1];
    const float* w_qkv     = (const float*)d_in[2];
    const float* w_o       = (const float*)d_in[3];
    float*       out       = (float*)d_out;

    unsigned short* qkv  = (unsigned short*)d_ws;                 // [4096, 6144] bf16
    unsigned short* attn = qkv + (size_t)4096 * 6144;             // [4096, 4096] bf16

    // 1) qkv = hidden @ w_qkv   (f32 in, bf16 out)
    gemm_wmma<true, true, false><<<dim3(6144 / 128, 4096 / 256), 256, 0, stream>>>(
        hidden, w_qkv, qkv, nullptr, 4096, 6144, 4096, 4096, 6144, 6144);

    // 2) partial RoPE in place + fold 1/sqrt(d) into Q
    rope_scale<<<4096, 256, 0, stream>>>(qkv, positions);

    // 3) causal flash attention (GQA 32q/8kv), bf16 WMMA, f32 accum
    fattn<<<dim3(4096 / 128, 32), 256, 8 * 4608 * sizeof(unsigned short), stream>>>(
        qkv, attn);

    // 4) out = attn @ w_o   (A via async-LDS copy, B f32->bf16, f32 out)
    gemm_wmma<false, true, true><<<dim3(4096 / 128, 4096 / 256), 256, 0, stream>>>(
        attn, w_o, nullptr, out, 4096, 4096, 4096, 4096, 4096, 4096);
}
